// GATModel_31336081392306
// MI455X (gfx1250) — matrix operand order repaired and checked by reading, not compiled, not run on hardware
//
#include <hip/hip_runtime.h>
#include <hip/hip_bf16.h>

// ---------------- problem constants ----------------
#define NN      50000
#define EIN     800000
#define EE      (EIN + NN)      // edges + self loops
#define INDIM   256
#define HIDC    64
#define HEADS   4
#define SLOPE   0.2f
#define EPS16   1e-16f

// ---------------- small helpers ----------------
typedef __attribute__((ext_vector_type(16))) __bf16        bf16x16;
typedef __attribute__((ext_vector_type(8)))  float         f32x8;
typedef __attribute__((ext_vector_type(4)))  float         f32x4;
typedef __attribute__((ext_vector_type(4)))  unsigned int  u32x4;

union FragU { bf16x16 v; u32x4 q[2]; };

__device__ __forceinline__ unsigned short f2bf(float f) {
    unsigned int u = __float_as_uint(f);
    unsigned int r = u + 0x7FFFu + ((u >> 16) & 1u);   // RNE
    return (unsigned short)(r >> 16);
}

// monotone float<->uint key so we can use integer atomicMax for segment-max
__device__ __forceinline__ unsigned fkey(float f) {
    unsigned b = __float_as_uint(f);
    return (b & 0x80000000u) ? ~b : (b | 0x80000000u);
}
__device__ __forceinline__ float unfkey(unsigned k) {
    unsigned b = (k & 0x80000000u) ? (k & 0x7FFFFFFFu) : ~k;
    return __uint_as_float(b);
}
#define KEY_NEG_INF 0x007FFFFFu   // fkey(-inf)

__device__ __forceinline__ void edge_sd(const int* __restrict__ ei, int i, int& s, int& d) {
    if (i < EIN) { s = ei[i]; d = ei[EIN + i]; }
    else         { s = i - EIN; d = i - EIN; }          // self loop
}

// ---------------- fill kernels ----------------
__global__ void fill_f32(float* p, int n, float v) {
    int i = blockIdx.x * blockDim.x + threadIdx.x;
    if (i < n) p[i] = v;
}
__global__ void fill_u32(unsigned* p, int n, unsigned v) {
    int i = blockIdx.x * blockDim.x + threadIdx.x;
    if (i < n) p[i] = v;
}

// ---------------- generic bf16 WMMA GEMM ----------------
// C[M,NCOL] = op( A[M,K] @ W[K,NCOL] (+ bias) ),  M = NN, tiles of 16 rows.
// W^T staged in LDS as bf16 ([NCOL][K]); A tile staged as bf16 ([16][K]).
template<int K, int NCOL, bool RELU, bool HASBIAS>
__global__ void gemm_wmma(const float* __restrict__ Ag,
                          const float* __restrict__ Wg,
                          const float* __restrict__ bias,
                          float* __restrict__ Cg)
{
    constexpr int WAVES = (NCOL >= 128) ? 8 : 4;
    constexpr int NT    = NCOL / (16 * WAVES);          // n-tiles per wave
    constexpr int TILES = NN / 16;                      // 3125, exact
    extern __shared__ unsigned short smem[];            // dynamic LDS
    unsigned short* lW = smem;                          // NCOL*K
    unsigned short* lA = smem + NCOL * K;               // 16*K

    const int tid      = threadIdx.x;
    const int nthreads = WAVES * 32;
    const int lane     = tid & 31;
    const int wave     = tid >> 5;
    const int g        = lane >> 4;                     // lane group (0/1)
    const int nl       = lane & 15;                     // n (and m) within tile

    // stage W^T into LDS once per block (coalesced global reads)
    for (int idx = tid; idx < K * NCOL; idx += nthreads) {
        int k = idx / NCOL, n = idx % NCOL;
        lW[n * K + k] = f2bf(Wg[idx]);
    }
    __syncthreads();

    for (int t = blockIdx.x; t < TILES; t += gridDim.x) {
        const int m0 = t * 16;
        // stage A tile (16 x K) as bf16
        for (int idx = tid; idx < 16 * K; idx += nthreads)
            lA[idx] = f2bf(Ag[m0 * K + idx]);
        // prefetch next tile's A rows
        int tn = t + gridDim.x;
        if (tn < TILES && lane == 0)
            __builtin_prefetch(&Ag[tn * 16 * K], 0, 1);
        __syncthreads();

        f32x8 acc[NT];
        #pragma unroll
        for (int nt = 0; nt < NT; ++nt) acc[nt] = (f32x8){0,0,0,0,0,0,0,0};

        #pragma unroll
        for (int kk = 0; kk < K / 32; ++kk) {
            // A fragment: lane holds row m=nl, K = kk*32 + {e/8*16 + g*8 + e%8}
            FragU af;
            const unsigned short* ap = lA + nl * K + kk * 32 + g * 8;
            af.q[0] = *(const u32x4*)(ap);
            af.q[1] = *(const u32x4*)(ap + 16);
            #pragma unroll
            for (int nt = 0; nt < NT; ++nt) {
                // B fragment: lane holds col n, K = kk*32 + {e/8*16 + g*8 + e%8}
                const int ncol = (wave * NT + nt) * 16 + nl;
                FragU bf;
                const unsigned short* bp = lW + ncol * K + kk * 32 + g * 8;
                bf.q[0] = *(const u32x4*)(bp);
                bf.q[1] = *(const u32x4*)(bp + 16);
                acc[nt] = __builtin_amdgcn_wmma_f32_16x16x32_bf16(
                    false, af.v, false, bf.v, (short)0, acc[nt], false, false);
            }
        }

        // epilogue: VGPR r holds M = r + 8*g, N = nl
        #pragma unroll
        for (int nt = 0; nt < NT; ++nt) {
            const int ncol0 = (wave * NT + nt) * 16;
            float bv = HASBIAS ? bias[ncol0 + nl] : 0.0f;
            #pragma unroll
            for (int r = 0; r < 8; ++r) {
                int row = m0 + r + 8 * g;
                float v = acc[nt][r] + bv;
                if (RELU) v = fmaxf(v, 0.0f);
                Cg[row * NCOL + ncol0 + nl] = v;
            }
        }
        __syncthreads();   // protect lA before restage
    }
}

// ---------------- attention score vectors ----------------
// s[n,h] = <h[n,h,:], a_src[h,:]>, d likewise. rowlen = H*64.
template<int H>
__global__ void attvec(const float* __restrict__ hx,
                       const float* __restrict__ asrc,
                       const float* __restrict__ adst,
                       float* __restrict__ s, float* __restrict__ d)
{
    int i = blockIdx.x * blockDim.x + threadIdx.x;
    if (i >= NN * H) return;
    int node = i / H, hd = i % H;
    const float* row = hx + node * (H * 64) + hd * 64;
    const float* av  = asrc + hd * 64;
    const float* dv  = adst + hd * 64;
    float ss = 0.f, dd = 0.f;
    #pragma unroll
    for (int c = 0; c < 64; c += 4) {
        f32x4 hv = *(const f32x4*)(row + c);
        f32x4 a  = *(const f32x4*)(av + c);
        f32x4 b  = *(const f32x4*)(dv + c);
        ss += hv[0]*a[0] + hv[1]*a[1] + hv[2]*a[2] + hv[3]*a[3];
        dd += hv[0]*b[0] + hv[1]*b[1] + hv[2]*b[2] + hv[3]*b[3];
    }
    s[i] = ss; d[i] = dd;
}

// ---------------- edge passes ----------------
template<int H>
__global__ void edge_max(const int* __restrict__ ei,
                         const float* __restrict__ s, const float* __restrict__ d,
                         unsigned* __restrict__ emaxk)
{
    int i = blockIdx.x * blockDim.x + threadIdx.x;
    if (i >= EE) return;
    int sn, dn; edge_sd(ei, i, sn, dn);
    #pragma unroll
    for (int h = 0; h < H; ++h) {
        float e = s[sn * H + h] + d[dn * H + h];
        float a = (e > 0.f) ? e : SLOPE * e;
        atomicMax(&emaxk[dn * H + h], fkey(a));
    }
}

template<int H>
__global__ void edge_exp(const int* __restrict__ ei,
                         const float* __restrict__ s, const float* __restrict__ d,
                         const unsigned* __restrict__ emaxk,
                         float* __restrict__ exw, float* __restrict__ denom)
{
    int i = blockIdx.x * blockDim.x + threadIdx.x;
    if (i >= EE) return;
    int sn, dn; edge_sd(ei, i, sn, dn);
    #pragma unroll
    for (int h = 0; h < H; ++h) {
        float e = s[sn * H + h] + d[dn * H + h];
        float a = (e > 0.f) ? e : SLOPE * e;
        float ex = __expf(a - unfkey(emaxk[dn * H + h]));
        exw[i * H + h] = ex;
        atomicAdd(&denom[dn * H + h], ex);
    }
}

// scatter: out[dst] += h[src] * alpha ;  H*16 threads per edge, float4 each
template<int H>
__global__ void edge_scatter(const int* __restrict__ ei,
                             const float* __restrict__ hx,
                             const float* __restrict__ exw,
                             const float* __restrict__ denom,
                             float* __restrict__ outx)
{
    constexpr int TPE = H * 16;
    long long gt = (long long)blockIdx.x * blockDim.x + threadIdx.x;
    if (gt >= (long long)EE * TPE) return;
    int i    = (int)(gt / TPE);
    int slot = (int)(gt % TPE);
    int c0   = slot * 4;
    int hd   = c0 >> 6;
    int sn, dn; edge_sd(ei, i, sn, dn);
    float alpha = exw[i * H + hd] / (denom[dn * H + hd] + EPS16);
    f32x4 v = *(const f32x4*)(hx + sn * (H * 64) + c0);
    float* ob = outx + dn * (H * 64) + c0;
    atomicAdd(ob + 0, v[0] * alpha);
    atomicAdd(ob + 1, v[1] * alpha);
    atomicAdd(ob + 2, v[2] * alpha);
    atomicAdd(ob + 3, v[3] * alpha);
}

// out = relu(out + bias) in place
__global__ void relu_bias(float* __restrict__ p, const float* __restrict__ bias, int n, int nc) {
    int i = blockIdx.x * blockDim.x + threadIdx.x;
    if (i < n * nc) p[i] = fmaxf(p[i] + bias[i % nc], 0.0f);
}

// row softmax over 64 cols, one wave per row (wave32: 2 elems/lane)
__global__ void softmax64(const float* __restrict__ logits, float* __restrict__ out) {
    int wave = threadIdx.x >> 5, lane = threadIdx.x & 31;
    int node = blockIdx.x * (blockDim.x >> 5) + wave;
    if (node >= NN) return;
    float x0 = logits[node * 64 + lane];
    float x1 = logits[node * 64 + 32 + lane];
    float m = fmaxf(x0, x1);
    #pragma unroll
    for (int o = 16; o > 0; o >>= 1) m = fmaxf(m, __shfl_xor(m, o, 32));
    float e0 = __expf(x0 - m), e1 = __expf(x1 - m);
    float sum = e0 + e1;
    #pragma unroll
    for (int o = 16; o > 0; o >>= 1) sum += __shfl_xor(sum, o, 32);
    float inv = 1.0f / sum;
    out[node * 64 + lane]      = e0 * inv;
    out[node * 64 + 32 + lane] = e1 * inv;
}

// ---------------- orchestration ----------------
static inline int ceil_div(long long a, int b) { return (int)((a + b - 1) / b); }

extern "C" void kernel_launch(void* const* d_in, const int* in_sizes, int n_in,
                              void* d_out, int out_size, void* d_ws, size_t ws_size,
                              hipStream_t stream)
{
    const float* x      = (const float*)d_in[0];
    const int*   ei     = (const int*)  d_in[1];
    const float* W1     = (const float*)d_in[2];
    const float* a_src1 = (const float*)d_in[3];
    const float* a_dst1 = (const float*)d_in[4];
    const float* b1     = (const float*)d_in[5];
    const float* W2     = (const float*)d_in[6];
    const float* a_src2 = (const float*)d_in[7];
    const float* a_dst2 = (const float*)d_in[8];
    const float* b2     = (const float*)d_in[9];
    const float* Wf1    = (const float*)d_in[10];
    const float* bf1    = (const float*)d_in[11];
    const float* Wf2    = (const float*)d_in[12];
    const float* bf2    = (const float*)d_in[13];
    float* outp = (float*)d_out;

    // workspace layout (floats)
    float*    A       = (float*)d_ws;                 // NN*256: h1, later h2/out2/f1/logits
    float*    B       = A + (size_t)NN * 256;         // NN*256: out1 -> hrelu1
    float*    s1      = B + (size_t)NN * 256;         // 4*NN
    float*    d1      = s1 + 4 * NN;                  // 4*NN
    unsigned* emaxk   = (unsigned*)(d1 + 4 * NN);     // 4*NN
    float*    denom   = (float*)(emaxk + 4 * NN);     // 4*NN
    float*    exw     = denom + 4 * NN;               // 4*EE
    // layer-2 reuse of region A
    float* h2     = A;
    float* out2   = A + (size_t)NN * 64;
    float* f1     = A + (size_t)NN * 128;
    float* logits = A + (size_t)NN * 192;

    const int BT = 256;

    // ---- layer 1 ----
    fill_f32<<<ceil_div((long long)NN * 256, BT), BT, 0, stream>>>(B, NN * 256, 0.0f);
    fill_f32<<<ceil_div(4 * NN, BT), BT, 0, stream>>>(denom, 4 * NN, 0.0f);
    fill_u32<<<ceil_div(4 * NN, BT), BT, 0, stream>>>(emaxk, 4 * NN, KEY_NEG_INF);

    {   // h1 = x @ W1
        size_t lds = (size_t)(256 * 256 + 16 * 256) * sizeof(unsigned short); // 136 KB
        gemm_wmma<256, 256, false, false><<<1024, 256, lds, stream>>>(x, W1, nullptr, A);
    }
    attvec<HEADS><<<ceil_div(NN * HEADS, BT), BT, 0, stream>>>(A, a_src1, a_dst1, s1, d1);
    edge_max<HEADS><<<ceil_div(EE, BT), BT, 0, stream>>>(ei, s1, d1, emaxk);
    edge_exp<HEADS><<<ceil_div(EE, BT), BT, 0, stream>>>(ei, s1, d1, emaxk, exw, denom);
    edge_scatter<HEADS><<<ceil_div((long long)EE * 64, BT), BT, 0, stream>>>(ei, A, exw, denom, B);
    relu_bias<<<ceil_div((long long)NN * 256, BT), BT, 0, stream>>>(B, b1, NN, 256);

    // ---- layer 2 ----
    {   // h2 = hrelu1 @ W2
        size_t lds = (size_t)(64 * 256 + 16 * 256) * sizeof(unsigned short);  // 40 KB
        gemm_wmma<256, 64, false, false><<<1024, 128, lds, stream>>>(B, W2, nullptr, h2);
    }
    attvec<1><<<ceil_div(NN, BT), BT, 0, stream>>>(h2, a_src2, a_dst2, s1, d1);
    fill_f32<<<ceil_div((long long)NN * 64, BT), BT, 0, stream>>>(out2, NN * 64, 0.0f);
    fill_f32<<<ceil_div(NN, BT), BT, 0, stream>>>(denom, NN, 0.0f);
    fill_u32<<<ceil_div(NN, BT), BT, 0, stream>>>(emaxk, NN, KEY_NEG_INF);
    edge_max<1><<<ceil_div(EE, BT), BT, 0, stream>>>(ei, s1, d1, emaxk);
    edge_exp<1><<<ceil_div(EE, BT), BT, 0, stream>>>(ei, s1, d1, emaxk, exw, denom);
    edge_scatter<1><<<ceil_div((long long)EE * 16, BT), BT, 0, stream>>>(ei, h2, exw, denom, out2);
    relu_bias<<<ceil_div((long long)NN * 64, BT), BT, 0, stream>>>(out2, b2, NN, 64);

    // ---- feed-forward + softmax ----
    {
        size_t lds = (size_t)(64 * 64 + 16 * 64) * sizeof(unsigned short);    // 10 KB
        gemm_wmma<64, 64, true,  true><<<2048, 128, lds, stream>>>(out2, Wf1, bf1, f1);
        gemm_wmma<64, 64, false, true><<<2048, 128, lds, stream>>>(f1,   Wf2, bf2, logits);
    }
    softmax64<<<ceil_div(NN, 8), 256, 0, stream>>>(logits, outp);
}
